// SparseMoE_33002528702983
// MI455X (gfx1250) — matrix operand order repaired and checked by reading, not compiled, not run on hardware
//
#include <hip/hip_runtime.h>
#include <hip/hip_bf16.h>
#include <math.h>

// Problem dims (match reference)
#define BB 4096
#define PP 16
#define DD 512
#define FF 2048
#define EE 8

#define SPB   4            // samples per block (M = 64 rows)
#define MM    (SPB * PP)   // 64
#define FB    256          // F-block for the h tile
#define NKEY  64           // pair keys (e1*8+e2)
#define NTHR  512          // 16 wave32s: low per-wave register pressure

typedef __attribute__((ext_vector_type(8)))  float  v8f;
typedef __attribute__((ext_vector_type(8)))  __bf16 bf16x8;
typedef __attribute__((ext_vector_type(16))) __bf16 bf16x16;

union BfPack {
  bf16x16 v;
  struct { bf16x8 lo; bf16x8 hi; } p;
};

// ---------------------------------------------------------------------------
// Router: noisy top-2 gates (dense, exact zeros elsewhere) + pair key +
// histogram of pair keys (for the counting sort that groups samples).
// ---------------------------------------------------------------------------
__global__ void moe_router_kernel(const float* __restrict__ aff,
                                  const float* __restrict__ noise,
                                  float* __restrict__ gates,
                                  int* __restrict__ keys,
                                  int* __restrict__ hist) {
  int b = blockIdx.x * blockDim.x + threadIdx.x;
  if (b >= BB) return;
  float v[EE];
#pragma unroll
  for (int e = 0; e < EE; ++e) {
    float a  = aff[b * EE + e];
    float sp = (a > 20.f) ? a : log1pf(__expf(a));   // stable softplus
    v[e] = a + noise[b * EE + e] * sp;
  }
  int i1 = 0; float v1 = v[0];
#pragma unroll
  for (int e = 1; e < EE; ++e) if (v[e] > v1) { v1 = v[e]; i1 = e; }
  int i2 = -1; float v2 = -INFINITY;
#pragma unroll
  for (int e = 0; e < EE; ++e) if (e != i1 && v[e] > v2) { v2 = v[e]; i2 = e; }
  float ez = __expf(v2 - v1);          // <= 1, safe
  float g1 = 1.f / (1.f + ez);
  float g2 = 1.f - g1;
#pragma unroll
  for (int e = 0; e < EE; ++e) gates[b * EE + e] = 0.f;
  gates[b * EE + i1] = g1;
  gates[b * EE + i2] = g2;

  int lo = i1 < i2 ? i1 : i2, hi = i1 < i2 ? i2 : i1;
  int key = lo * EE + hi;              // unordered pair id
  keys[b] = key;
  atomicAdd(&hist[key], 1);
}

__global__ void moe_zero_hist(int* __restrict__ hist) {
  if (threadIdx.x < NKEY) hist[threadIdx.x] = 0;
}

__global__ void moe_prefix(const int* __restrict__ hist, int* __restrict__ cursor) {
  if (threadIdx.x == 0) {
    int run = 0;
    for (int k = 0; k < NKEY; ++k) { cursor[k] = run; run += hist[k]; }
  }
}

__global__ void moe_scatter(const int* __restrict__ keys, int* __restrict__ cursor,
                            int* __restrict__ order) {
  int b = blockIdx.x * blockDim.x + threadIdx.x;
  if (b >= BB) return;
  int pos = atomicAdd(&cursor[keys[b]], 1);  // order within a bin is irrelevant
  order[pos] = b;
}

// ---------------------------------------------------------------------------
// Transpose + fp32->bf16 convert: dst[z][c][r] = (bf16) src[z][r][c].
// Gives the WMMA B-operand a K-contiguous layout (contiguous 16B loads).
// ---------------------------------------------------------------------------
__global__ void transpose_cvt_kernel(const float* __restrict__ src,
                                     __bf16* __restrict__ dst,
                                     int rows, int cols) {
  __shared__ float tile[32][33];
  size_t slice = (size_t)rows * cols;
  const float* s = src + (size_t)blockIdx.z * slice;
  __bf16*      d = dst + (size_t)blockIdx.z * slice;
  int c0 = blockIdx.x * 32, r0 = blockIdx.y * 32;
  int tx = threadIdx.x & 31, ty = threadIdx.x >> 5;   // 32 x 8
#pragma unroll
  for (int i = 0; i < 32; i += 8)
    tile[ty + i][tx] = s[(size_t)(r0 + ty + i) * cols + (c0 + tx)];
  __syncthreads();
#pragma unroll
  for (int i = 0; i < 32; i += 8)
    d[(size_t)(c0 + ty + i) * rows + (r0 + tx)] = (__bf16)tile[tx][ty + i];
}

// ---------------------------------------------------------------------------
// Fused MoE kernel, pair-sorted: 4 samples per block (M = 64 = 4 WMMA M-tiles,
// one sample per M-tile), 16 wave32s. For each expert used by any resident
// sample:
//   GEMM1 (K=512) -> h = gate * gelu(...) in LDS (bf16, F blocked at 256)
//   GEMM2 (K=256 per F-block) accumulates straight into persistent yacc via
//   the WMMA C operand. gate==0 rows contribute exactly +0.
// Per-wave registers: yacc 64 + acc 32 + operands ~48 -> no spills.
// ---------------------------------------------------------------------------
#define XPITCH (DD + 8)   // 520 bf16: 260-dword row stride, conflict-free
#define HPITCH (FB + 8)   // 264 bf16: 132-dword row stride, conflict-free

// tanh-GELU via sigmoid identity: 0.5*(1+tanh(u)) == 1/(1+exp(-2u)).
// exp2/rcp lower to single v_exp_f32 / v_rcp_f32; bf16 destination, so
// ~22-bit rcp accuracy is more than enough.
__device__ __forceinline__ float fast_gelu(float z) {
  const float t = -(2.3022082f + 0.1029482f * z * z) * z;  // -2u * log2(e)
  return z * __builtin_amdgcn_rcpf(1.f + __builtin_amdgcn_exp2f(t));
}

__global__ __launch_bounds__(NTHR)
void moe_expert_kernel(const float* __restrict__ x,       // [B][P][D]
                       const float* __restrict__ gates,   // [B][E]
                       const int* __restrict__ order,     // [B] pair-sorted
                       const __bf16* __restrict__ W1T,    // [E][F][D]
                       const float* __restrict__ b1,      // [E][F]
                       const __bf16* __restrict__ W2T,    // [E][D][F]
                       const float* __restrict__ b2,      // [E][D]
                       float* __restrict__ out) {         // [B][P][D]
  __shared__ __align__(16) __bf16 xs[MM * XPITCH];  // 66,560 B
  __shared__ __align__(16) __bf16 hs[MM * HPITCH];  // 33,792 B

  const int tid  = threadIdx.x;
  const int lane = tid & 31;
  const int wave = tid >> 5;    // 0..15
  const int hi   = lane >> 4;
  const int m16  = lane & 15;

  int sm[SPB];
#pragma unroll
  for (int mt = 0; mt < SPB; ++mt) sm[mt] = order[blockIdx.x * SPB + mt];

  // Stage 4 gathered samples (64 x 512 fp32) into LDS as bf16 (A layout).
  for (int i = tid; i < MM * DD / 4; i += NTHR) {
    const int idx = i * 4;
    const int row = idx / DD, c = idx % DD;        // row 0..63
    const int mt = row >> 4, m = row & 15;
    const float4 f =
        ((const float4*)(x + (size_t)sm[mt] * (PP * DD)))[(m * DD + c) >> 2];
    __bf16* dst = &xs[row * XPITCH + c];
    dst[0] = (__bf16)f.x; dst[1] = (__bf16)f.y;
    dst[2] = (__bf16)f.z; dst[3] = (__bf16)f.w;
  }

  v8f yacc[SPB][2] = {};   // wave owns 2 D-tiles x 4 M-tiles: 64 VGPRs
  __syncthreads();

  for (int e = 0; e < EE; ++e) {
    float gmt[SPB];
    bool any = false;
#pragma unroll
    for (int mt = 0; mt < SPB; ++mt) {
      gmt[mt] = gates[sm[mt] * EE + e];
      any = any || (gmt[mt] != 0.f);
    }
    if (!any) continue;                    // block-uniform branch
    const __bf16* w1 = W1T + (size_t)e * FF * DD;
    const __bf16* w2 = W2T + (size_t)e * DD * FF;

    for (int fb = 0; fb < FF; fb += FB) {
      // ---- GEMM1: h[:, fb:fb+FB] = gate * gelu(x @ W1 + b1), K = 512 ----
      {
        const int ncol = fb + wave * 16 + m16;     // one N-tile per wave
        const __bf16* wcol = w1 + (size_t)ncol * DD + hi * 16;
        v8f acc[SPB] = {};
#pragma unroll 4
        for (int k0 = 0; k0 < DD; k0 += 32) {
          BfPack bm;                                // B reused across 4 M-tiles
          bm.p.lo = *(const bf16x8*)(wcol + k0);
          bm.p.hi = *(const bf16x8*)(wcol + k0 + 8);
#pragma unroll
          for (int mt = 0; mt < SPB; ++mt) {
            BfPack a;
            const __bf16* ar = &xs[(mt * 16 + m16) * XPITCH + k0 + hi * 16];
            a.p.lo = *(const bf16x8*)ar;
            a.p.hi = *(const bf16x8*)(ar + 8);
            acc[mt] = __builtin_amdgcn_wmma_f32_16x16x32_bf16(
                false, a.v, false, bm.v, (short)0, acc[mt], false, false);
          }
        }
        const float bias = b1[e * FF + ncol];
#pragma unroll
        for (int mt = 0; mt < SPB; ++mt) {
#pragma unroll
          for (int r = 0; r < 8; ++r) {            // row = r + 8*hi
            const float gel = fast_gelu(acc[mt][r] + bias);
            hs[(mt * 16 + r + 8 * hi) * HPITCH + wave * 16 + m16] =
                (__bf16)(gmt[mt] * gel);           // gate folded into h
          }
        }
      }
      __syncthreads();

      // ---- GEMM2 partial: yacc += h_blk @ W2[fb:fb+FB, :], K = 256 ----
#pragma unroll
      for (int nn = 0; nn < 2; ++nn) {
        const int nt   = wave + nn * 16;           // D tile 0..31
        const int ncol = nt * 16 + m16;
        const __bf16* wcol = w2 + (size_t)ncol * FF + fb + hi * 16;
#pragma unroll 4
        for (int k0 = 0; k0 < FB; k0 += 32) {      // 8 K-chunks
          BfPack bm;
          bm.p.lo = *(const bf16x8*)(wcol + k0);
          bm.p.hi = *(const bf16x8*)(wcol + k0 + 8);
#pragma unroll
          for (int mt = 0; mt < SPB; ++mt) {
            BfPack a;
            const __bf16* ar = &hs[(mt * 16 + m16) * HPITCH + k0 + hi * 16];
            a.p.lo = *(const bf16x8*)ar;
            a.p.hi = *(const bf16x8*)(ar + 8);
            yacc[mt][nn] = __builtin_amdgcn_wmma_f32_16x16x32_bf16(
                false, a.v, false, bm.v, (short)0, yacc[mt][nn], false, false);
          }
        }
      }
      __syncthreads();   // hs reused by next F-block / expert
    }

    // gate-weighted per-expert output bias b2 (zero-gate rows add 0)
#pragma unroll
    for (int nn = 0; nn < 2; ++nn) {
      const int ncol = (wave + nn * 16) * 16 + m16;
      const float b2v = b2[e * DD + ncol];
#pragma unroll
      for (int mt = 0; mt < SPB; ++mt)
#pragma unroll
        for (int r = 0; r < 8; ++r) yacc[mt][nn][r] += gmt[mt] * b2v;
    }
  }

  // Write out: block owns its samples entirely (no atomics).
#pragma unroll
  for (int mt = 0; mt < SPB; ++mt) {
    float* ob = out + (size_t)sm[mt] * (PP * DD);
#pragma unroll
    for (int nn = 0; nn < 2; ++nn) {
      const int ncol = (wave + nn * 16) * 16 + m16;
#pragma unroll
      for (int r = 0; r < 8; ++r)
        ob[(r + 8 * hi) * DD + ncol] = yacc[mt][nn][r];
    }
  }
}

// ---------------------------------------------------------------------------
extern "C" void kernel_launch(void* const* d_in, const int* in_sizes, int n_in,
                              void* d_out, int out_size, void* d_ws, size_t ws_size,
                              hipStream_t stream) {
  (void)in_sizes; (void)n_in; (void)out_size; (void)ws_size;
  const float* patch_x  = (const float*)d_in[0];
  // d_in[1] = patch_embedding: only defines output shape; unused.
  const float* affinity = (const float*)d_in[2];
  const float* noise    = (const float*)d_in[3];
  const float* W1       = (const float*)d_in[4];
  const float* b1       = (const float*)d_in[5];
  const float* W2       = (const float*)d_in[6];
  const float* b2       = (const float*)d_in[7];
  float*       out      = (float*)d_out;

  // Workspace layout (all 256B aligned):
  char* ws = (char*)d_ws;
  size_t off = 0;
  float* gates  = (float*)(ws + off); off += (size_t)BB * EE * sizeof(float);
  int*   keys   = (int*)  (ws + off); off += (size_t)BB * sizeof(int);
  int*   hist   = (int*)  (ws + off); off += 256;
  int*   cursor = (int*)  (ws + off); off += 256;
  int*   order  = (int*)  (ws + off); off += (size_t)BB * sizeof(int);
  off = (off + 255) & ~(size_t)255;
  __bf16* W1T   = (__bf16*)(ws + off); off += (size_t)EE * FF * DD * sizeof(__bf16);
  __bf16* W2T   = (__bf16*)(ws + off);

  moe_zero_hist<<<1, 64, 0, stream>>>(hist);
  moe_router_kernel<<<BB / 256, 256, 0, stream>>>(affinity, noise, gates, keys, hist);
  moe_prefix<<<1, 1, 0, stream>>>(hist, cursor);
  moe_scatter<<<BB / 256, 256, 0, stream>>>(keys, cursor, order);

  dim3 g1(FF / 32, DD / 32, EE);   // W1 [E][D][F] -> W1T [E][F][D]
  transpose_cvt_kernel<<<g1, 256, 0, stream>>>(W1, W1T, DD, FF);
  dim3 g2(DD / 32, FF / 32, EE);   // W2 [E][F][D] -> W2T [E][D][F]
  transpose_cvt_kernel<<<g2, 256, 0, stream>>>(W2, W2T, FF, DD);

  moe_expert_kernel<<<BB / SPB, NTHR, 0, stream>>>(patch_x, gates, order,
                                                   W1T, b1, W2T, b2, out);
}
